// unit_gcn_55722905698718
// MI455X (gfx1250) — matrix operand order repaired and checked by reading, not compiled, not run on hardware
//
#include <hip/hip_runtime.h>

// ---------------------------------------------------------------------------
// unit_gcn for MI455X (gfx1250): bf16 WMMA GEMMs + f32 BN stats, 2-pass.
//   Ae = PA + A*edge (25x25, zero-padded to 32x32, bf16, stored transposed)
//   W  = W0 + W1     (64x64, bf16)   [biases cancel through training-mode BN]
//   pass1: per-tile y = W * (x * Ae); accumulate per-channel sum / sum^2
//   mid:   scale = gamma/sqrt(var+eps), shift = beta - mean*scale
//   pass2: recompute y tile; out = relu(y*scale + shift + x)
// All WMMA fragments are built from contiguous 16B LDS vector accesses.
// ---------------------------------------------------------------------------

typedef __attribute__((ext_vector_type(16))) __bf16 v16bf;
typedef __attribute__((ext_vector_type(8)))  __bf16 v8bf;
typedef __attribute__((ext_vector_type(8)))  float  v8f;

#define N_   128
#define C_   64
#define T_   300
#define V_   25
#define TT   4            // timesteps per workgroup tile
#define NCHUNK (T_ / TT)  // 75

// padded LDS strides (bf16 elements; keep 16B alignment, spread banks)
#define VS   40           // xs row stride   (v dim, 32 used)
#define AS   40           // aeT row stride  (v dim, 32 used)
#define WS   72           // wb row stride   (c dim, 64 used)
#define KS   80           // zsT col stride  (c dim, 64 used)

// workspace layout (bytes)
#define WS_AE    0        // aeT: 32*32 bf16 = 2048   (aeT[w][v])
#define WS_W     2048     // 64*64 bf16 = 8192
#define WS_SUM   10240    // 64 f32
#define WS_SQ    10496    // 64 f32
#define WS_SCALE 10752    // 64 f32
#define WS_SHIFT 11008    // 64 f32

// -------------------------------------------------------------- prep kernel
__global__ __launch_bounds__(256) void gcn_prep(
    const float* __restrict__ A, const float* __restrict__ PA,
    const float* __restrict__ E, const float* __restrict__ W0,
    const float* __restrict__ W1, char* __restrict__ ws)
{
    int tid = threadIdx.x;
    __bf16* aet = (__bf16*)(ws + WS_AE);
    __bf16* wb  = (__bf16*)(ws + WS_W);
    float*  st  = (float*)(ws + WS_SUM);
    for (int i = tid; i < 1024; i += 256) {
        int w = i >> 5, v = i & 31;                 // transposed: [w][v]
        float val = (v < V_ && w < V_)
                        ? (PA[v * V_ + w] + A[v * V_ + w] * E[v * V_ + w])
                        : 0.0f;
        aet[i] = (__bf16)val;
    }
    for (int i = tid; i < 4096; i += 256)
        wb[i] = (__bf16)(W0[i] + W1[i]);
    for (int i = tid; i < 128; i += 256)            // zero sum + sumsq
        st[i] = 0.0f;
}

// ---------------------------------------------------------- finalize kernel
__global__ void gcn_finalize(const float* __restrict__ gamma,
                             const float* __restrict__ beta,
                             char* __restrict__ ws)
{
    int o = threadIdx.x;
    if (o < C_) {
        const float cnt = (float)N_ * (float)T_ * (float)V_;
        float s = ((float*)(ws + WS_SUM))[o];
        float q = ((float*)(ws + WS_SQ))[o];
        float mean = s / cnt;
        float var  = q / cnt - mean * mean;
        float sc   = gamma[o] * rsqrtf(var + 1e-5f);
        ((float*)(ws + WS_SCALE))[o] = sc;
        ((float*)(ws + WS_SHIFT))[o] = beta[o] - mean * sc;
    }
}

__device__ __forceinline__ v16bf joinbf(v8bf lo, v8bf hi) {
    return __builtin_shufflevector(lo, hi, 0, 1, 2, 3, 4, 5, 6, 7,
                                   8, 9, 10, 11, 12, 13, 14, 15);
}

// ---------------------------------------------------------------- main GEMM
// MODE 0: accumulate BN statistics.  MODE 1: apply BN + residual + relu.
template <int MODE>
__global__ __launch_bounds__(256) void gcn_main(
    const float* __restrict__ x, char* __restrict__ ws,
    float* __restrict__ out)
{
    // LDS: 53,248 bytes total
    __shared__ __align__(32) __bf16 xs[256 * VS];  // x tile, row R=t'*64+c, col v
    __shared__ __align__(32) __bf16 aet[32 * AS];  // Ae^T: [w][v]
    __shared__ __align__(32) __bf16 wb[64 * WS];   // W_eff: [o][c]
    __shared__ __align__(32) __bf16 zst[128 * KS]; // z^T: [col=t'*32+w][c]
    __shared__ float red0[64];                     // MODE0: sum    MODE1: scale
    __shared__ float red1[64];                     // MODE0: sumsq  MODE1: shift

    const int tid = threadIdx.x;
    const int n   = blockIdx.y;
    const int t0  = blockIdx.x * TT;

    const __bf16* aeg = (const __bf16*)(ws + WS_AE);
    const __bf16* wg  = (const __bf16*)(ws + WS_W);

    for (int i = tid; i < 1024; i += 256) aet[(i >> 5) * AS + (i & 31)] = aeg[i];
    for (int i = tid; i < 4096; i += 256) wb[(i >> 6) * WS + (i & 63)] = wg[i];
    if (tid < 64) {
        if (MODE == 0) { red0[tid] = 0.0f; red1[tid] = 0.0f; }
        else {
            red0[tid] = ((const float*)(ws + WS_SCALE))[tid];
            red1[tid] = ((const float*)(ws + WS_SHIFT))[tid];
        }
    }
    // x tile -> bf16 LDS, zero-padded v>=25. One bf16 pair per iteration.
    const float* xn = x + (size_t)n * C_ * T_ * V_;
    for (int i = tid; i < 256 * 16; i += 256) {
        int row = i >> 4, p = i & 15;          // row = t'*64 + c, v pair = 2p
        int tp = row >> 6, c = row & 63, v = 2 * p;
        const float* src = xn + c * T_ * V_ + (t0 + tp) * V_;
        float f0 = (v     < V_) ? src[v]     : 0.0f;
        float f1 = (v + 1 < V_) ? src[v + 1] : 0.0f;
        __bf16* d = &xs[row * VS + v];
        d[0] = (__bf16)f0;
        d[1] = (__bf16)f1;
    }
    __syncthreads();

    const int lane = tid & 31;
    const int wv   = tid >> 5;
    const int m    = lane & 15;
    const int half = lane >> 4;

    // ---- stage 1: z[(t',c), w] = sum_v x * Ae  (K = v padded to 32) ----
#pragma unroll
    for (int ti = 0; ti < 4; ++ti) {
        int tIdx = wv * 4 + ti;            // 0..31
        int mt = tIdx >> 1, nt = tIdx & 1; // 16 row tiles x 2 w tiles
        int R = mt * 16 + m;
        // A frag: two contiguous 8xbf16 chunks of row R
        const __bf16* ap = &xs[R * VS];
        v16bf a = joinbf(*(const v8bf*)(ap + 8 * half),
                         *(const v8bf*)(ap + 16 + 8 * half));
        // B frag: column w = nt*16+m of Ae -> contiguous in aeT
        const __bf16* bp = &aet[(nt * 16 + m) * AS];
        v16bf b = joinbf(*(const v8bf*)(bp + 8 * half),
                         *(const v8bf*)(bp + 16 + 8 * half));
        v8f c = {};
        c = __builtin_amdgcn_wmma_f32_16x16x32_bf16(
                false, a, false, b, (short)0, c, false, false);
        // D: lane holds 8 consecutive channels of one z column -> packed store
        int tp  = mt >> 2;
        int cb  = (mt & 3) * 16 + 8 * half;
        int col = tp * 32 + nt * 16 + m;
        v8bf zrow;
#pragma unroll
        for (int r = 0; r < 8; ++r) zrow[r] = (__bf16)c[r];
        *(v8bf*)&zst[col * KS + cb] = zrow;
    }
    __syncthreads();

    // ---- stage 2: y[o, (t',w)] = sum_c W[o,c] * z[c, (t',w)]  (K = 64) ----
    const int ot = wv >> 1;                // constant per wave
    float sum0[8], sum1[8];
    if (MODE == 0) {
#pragma unroll
        for (int r = 0; r < 8; ++r) { sum0[r] = 0.0f; sum1[r] = 0.0f; }
    }
#pragma unroll
    for (int ti = 0; ti < 4; ++ti) {
        int colt = (wv & 1) * 4 + ti;      // 8 col tiles
        v8f acc = {};
#pragma unroll
        for (int ks = 0; ks < 2; ++ks) {
            const __bf16* ap = &wb[(ot * 16 + m) * WS + ks * 32];
            v16bf a = joinbf(*(const v8bf*)(ap + 8 * half),
                             *(const v8bf*)(ap + 16 + 8 * half));
            const __bf16* bp = &zst[(colt * 16 + m) * KS + ks * 32];
            v16bf b = joinbf(*(const v8bf*)(bp + 8 * half),
                             *(const v8bf*)(bp + 16 + 8 * half));
            acc = __builtin_amdgcn_wmma_f32_16x16x32_bf16(
                      false, a, false, b, (short)0, acc, false, false);
        }
        if (MODE == 0) {
            // padded w>=25 columns are exactly zero -> safe to include
#pragma unroll
            for (int r = 0; r < 8; ++r) {
                sum0[r] += acc[r];
                sum1[r] += acc[r] * acc[r];
            }
        } else {
            int col = colt * 16 + m;
            int tp  = col >> 5, w = col & 31;
            if (w < V_) {
#pragma unroll
                for (int r = 0; r < 8; ++r) {
                    int o = ot * 16 + r + 8 * half;
                    size_t gi = (((size_t)n * C_ + o) * T_ + (t0 + tp)) * V_ + w;
                    float yv = acc[r] * red0[o] + red1[o] + x[gi]; // residual
                    out[gi] = fmaxf(yv, 0.0f);
                }
            }
        }
    }
    if (MODE == 0) {
#pragma unroll
        for (int r = 0; r < 8; ++r) {
            int o = ot * 16 + r + 8 * half;
            atomicAdd(&red0[o], sum0[r]);
            atomicAdd(&red1[o], sum1[r]);
        }
        __syncthreads();
        if (tid < 64) {
            atomicAdd(&((float*)(ws + WS_SUM))[tid], red0[tid]);
            atomicAdd(&((float*)(ws + WS_SQ))[tid],  red1[tid]);
        }
    }
}

// ------------------------------------------------------------------ launch
extern "C" void kernel_launch(void* const* d_in, const int* in_sizes, int n_in,
                              void* d_out, int out_size, void* d_ws,
                              size_t ws_size, hipStream_t stream)
{
    const float* x     = (const float*)d_in[0];
    const float* A     = (const float*)d_in[1];
    const float* PA    = (const float*)d_in[2];
    const float* E     = (const float*)d_in[3];
    const float* W0    = (const float*)d_in[4];
    /* d_in[5] = b0, d_in[7] = b1: cancel exactly through training-mode BN */
    const float* W1    = (const float*)d_in[6];
    const float* gamma = (const float*)d_in[8];
    const float* beta  = (const float*)d_in[9];
    float* out = (float*)d_out;
    char*  ws  = (char*)d_ws;

    gcn_prep<<<1, 256, 0, stream>>>(A, PA, E, W0, W1, ws);
    gcn_main<0><<<dim3(NCHUNK, N_), 256, 0, stream>>>(x, ws, nullptr);
    gcn_finalize<<<1, 64, 0, stream>>>(gamma, beta, ws);
    gcn_main<1><<<dim3(NCHUNK, N_), 256, 0, stream>>>(x, ws, out);
}
